// CrossEntropyLoss_74423193305825
// MI455X (gfx1250) — matrix operand order repaired and checked by reading, not compiled, not run on hardware
//
#include <hip/hip_runtime.h>

typedef __attribute__((ext_vector_type(16))) _Float16 v16h;
typedef __attribute__((ext_vector_type(8)))  float    v8f;

#define B_    4
#define S_    2048
#define H_    4096
#define V_    32000
#define NROW  (B_*(S_-1))        // 8188 valid rows
#define NPAD  8192               // padded row count (multiple of 32)
#define NCHUNK 5                 // V split; 6400 cols per chunk (52MB f16, fits L2)
#define VC    (V_/NCHUNK)        // 6400
#define MBLK  32                 // rows per CTA
#define NWAVE 16                 // 512 threads: 16 waves = 4 per SIMD32 (latency hiding)
#define CSTEP (NWAVE*16)         // 256 columns per step
#define NSTEP (VC/CSTEP)         // 25

// ---- workspace layout (bytes) ----
#define OFF_HS 0ull
#define OFF_W  (OFF_HS + (unsigned long long)NPAD*H_*2ull)            // f16 hidden panel
#define OFF_PM (OFF_W  + (unsigned long long)V_*H_*2ull)              // f16 weight
#define OFF_PS (OFF_PM + (unsigned long long)NPAD*NCHUNK*4ull)        // per-(row,chunk) max
#define OFF_LB (OFF_PS + (unsigned long long)NPAD*NCHUNK*4ull)        // per-(row,chunk) sumexp
// total ~330 MB workspace assumed available

#define APAN_BYTES ((size_t)MBLK*H_*sizeof(_Float16))                 // 256 KB

// ---------------- fp32 -> f16 conversion (with causal shift + zero pad) -------------
__global__ void ce_cvt_hidden(const float* __restrict__ hs, _Float16* __restrict__ hs16) {
    long idx = (long)blockIdx.x * blockDim.x + threadIdx.x;     // one thread per 4 elems
    if (idx >= (long)NPAD * (H_/4)) return;
    long n = idx / (H_/4);
    int  h = (int)(idx % (H_/4)) * 4;
    float4 v = make_float4(0.f, 0.f, 0.f, 0.f);
    if (n < NROW) {
        long bb = n / (S_-1), t = n % (S_-1);
        v = *(const float4*)(hs + (bb*S_ + t) * (long)H_ + h);
    }
    union { _Float16 x[4]; uint2 u; } p;
    p.x[0] = (_Float16)v.x; p.x[1] = (_Float16)v.y;
    p.x[2] = (_Float16)v.z; p.x[3] = (_Float16)v.w;
    *(uint2*)(hs16 + n * (long)H_ + h) = p.u;
}

__global__ void ce_cvt_weight(const float* __restrict__ w, _Float16* __restrict__ w16) {
    long idx = (long)blockIdx.x * blockDim.x + threadIdx.x;     // one thread per 4 elems
    if (idx >= (long)V_ * (H_/4)) return;
    float4 v = *(const float4*)(w + idx*4);
    union { _Float16 x[4]; uint2 u; } p;
    p.x[0] = (_Float16)v.x; p.x[1] = (_Float16)v.y;
    p.x[2] = (_Float16)v.z; p.x[3] = (_Float16)v.w;
    *(uint2*)(w16 + idx*4) = p.u;
}

// 16-lane-half butterfly reductions (rows of a WMMA C tile live in 16-lane halves)
__device__ __forceinline__ float hmax16(float v) {
    v = fmaxf(v, __shfl_xor(v, 1, 16));
    v = fmaxf(v, __shfl_xor(v, 2, 16));
    v = fmaxf(v, __shfl_xor(v, 4, 16));
    v = fmaxf(v, __shfl_xor(v, 8, 16));
    return v;
}
__device__ __forceinline__ float hsum16(float v) {
    v += __shfl_xor(v, 1, 16);
    v += __shfl_xor(v, 2, 16);
    v += __shfl_xor(v, 4, 16);
    v += __shfl_xor(v, 8, 16);
    return v;
}

// ---------------- fused GEMM + online softmax over a V chunk -------------------------
// grid: (NPAD/MBLK, NCHUNK), block: 512 threads = 16 waves, each wave = 1 col-tile
// and BOTH row-tiles (2 accumulators, B fragment reused by 2 WMMAs).
// dynamic LDS: A panel 32x4096 f16 (256KB) + per-(row,wave) (m,s) partials + labels
__global__ __launch_bounds__(512) void ce_gemm(
        const _Float16* __restrict__ hs16, const _Float16* __restrict__ w16,
        const int* __restrict__ labels,
        float* __restrict__ pm, float* __restrict__ ps, float* __restrict__ lab) {
    extern __shared__ char smem[];
    _Float16* Apan  = (_Float16*)smem;                          // [32][4096] f16
    float*    partM = (float*)(smem + APAN_BYTES);              // [32][NWAVE]
    float*    partS = partM + MBLK*NWAVE;                       // [32][NWAVE]
    int*      lblS  = (int*)(partS + MBLK*NWAVE);               // [32]

    const int tid    = threadIdx.x;
    const int rowblk = blockIdx.x;
    const int chunk  = blockIdx.y;
    const long gRow0 = (long)rowblk * MBLK;

    // ---- stage whole A panel into LDS (contiguous 256KB copy) ----
    {
        const uint4* src = (const uint4*)(hs16 + (size_t)rowblk * MBLK * H_);
        uint4* dst = (uint4*)Apan;
        #pragma unroll 4
        for (int i = tid; i < MBLK*H_/8; i += 512) dst[i] = src[i];
    }
    // ---- stage the 32 (shifted) labels ----
    if (tid < MBLK) {
        long n = gRow0 + tid;
        int lb = -1;
        if (n < NROW) {
            long bb = n / (S_-1), t = n % (S_-1);
            lb = labels[bb*S_ + t + 1];               // causal shift
        }
        lblS[tid] = lb;
    }
    __syncthreads();

    const int wave = tid >> 5, lane = tid & 31;       // wave == col-tile index (0..15)
    const int hi = (lane >> 4) & 1, l16 = lane & 15;
    const _Float16* aBase0 = Apan + (     l16) * H_ + hi*8;   // row-tile 0 (rows 0..15)
    const _Float16* aBase1 = Apan + (16 + l16) * H_ + hi*8;   // row-tile 1 (rows 16..31)

    // per-row online-softmax running state (owned by threads 0..31 == rows)
    float m = -3.0e38f, s = 0.0f;

    for (int step = 0; step < NSTEP; ++step) {
        const int col0 = chunk*VC + step*CSTEP;
        // B: column (col0 + wave*16 + l16) of W^T == row of W, contiguous along K
        const _Float16* bBase = w16 + (size_t)(col0 + wave*16 + l16) * H_ + hi*8;

        v8f c0 = {0.f,0.f,0.f,0.f,0.f,0.f,0.f,0.f};
        v8f c1 = {0.f,0.f,0.f,0.f,0.f,0.f,0.f,0.f};
        union V32 { v16h h; uint4 u[2]; };
        #pragma unroll 4
        for (int k = 0; k < H_; k += 32) {
            V32 a0, a1, b;
            b.u[0]  = *(const uint4*)(bBase  + k);       // K k..k+7   (lane-relative)
            b.u[1]  = *(const uint4*)(bBase  + k + 16);  // K k+16..k+23
            a0.u[0] = *(const uint4*)(aBase0 + k);
            a0.u[1] = *(const uint4*)(aBase0 + k + 16);
            a1.u[0] = *(const uint4*)(aBase1 + k);
            a1.u[1] = *(const uint4*)(aBase1 + k + 16);
            c0 = __builtin_amdgcn_wmma_f32_16x16x32_f16(
                     false, a0.h, false, b.h, (short)0, c0, false, false);
            c1 = __builtin_amdgcn_wmma_f32_16x16x32_f16(
                     false, a1.h, false, b.h, (short)0, c1, false, false);
        }

        // ---- in-register softmax partials: per-row (max, sumexp) over this wave's
        //      16 columns, via 16-lane butterflies (all 512 lanes busy) ----
        const int colL = col0 + wave*16 + l16;            // this lane's column
        #pragma unroll
        for (int i = 0; i < 8; ++i) {
            {   // row-tile 0: row = i + hi*8
                float x  = c0[i];
                float mv = hmax16(x);
                float sv = hsum16(__expf(x - mv));
                int row = i + hi*8;
                if (l16 == 0) { partM[row*NWAVE + wave] = mv; partS[row*NWAVE + wave] = sv; }
                if (lblS[row] == colL) lab[gRow0 + row] = x;
            }
            {   // row-tile 1: row = 16 + i + hi*8
                float x  = c1[i];
                float mv = hmax16(x);
                float sv = hsum16(__expf(x - mv));
                int row = 16 + i + hi*8;
                if (l16 == 0) { partM[row*NWAVE + wave] = mv; partS[row*NWAVE + wave] = sv; }
                if (lblS[row] == colL) lab[gRow0 + row] = x;
            }
        }
        __syncthreads();

        // ---- merge the NWAVE per-wave partials into the running (m, s) ----
        if (tid < MBLK) {
            float bm = m;
            #pragma unroll
            for (int wv = 0; wv < NWAVE; ++wv) bm = fmaxf(bm, partM[tid*NWAVE + wv]);
            float add = 0.f;
            #pragma unroll
            for (int wv = 0; wv < NWAVE; ++wv)
                add += partS[tid*NWAVE + wv] * __expf(partM[tid*NWAVE + wv] - bm);
            s = s * __expf(m - bm) + add;
            m = bm;
        }
        __syncthreads();   // partM/partS reusable next step
    }

    if (tid < MBLK) {
        pm[(gRow0 + tid)*NCHUNK + chunk] = m;
        ps[(gRow0 + tid)*NCHUNK + chunk] = s;
    }
}

// ---------------- combine chunk partials, mean NLL ----------------------------------
__global__ __launch_bounds__(256) void ce_final(
        const float* __restrict__ pm, const float* __restrict__ ps,
        const float* __restrict__ lab, const int* __restrict__ labels,
        float* __restrict__ out) {
    __shared__ float sSum[256];
    __shared__ float sCnt[256];
    float acc = 0.f, cnt = 0.f;
    for (int n = threadIdx.x; n < NROW; n += 256) {
        long bb = n / (S_-1), t = n % (S_-1);
        int lbl = labels[bb*S_ + t + 1];
        if (lbl == -100) continue;                    // IGNORE_INDEX
        float M = pm[n*NCHUNK + 0];
        #pragma unroll
        for (int c = 1; c < NCHUNK; ++c) M = fmaxf(M, pm[n*NCHUNK + c]);
        float S = 0.f;
        #pragma unroll
        for (int c = 0; c < NCHUNK; ++c) S += ps[n*NCHUNK + c] * __expf(pm[n*NCHUNK + c] - M);
        acc += -(lab[n] - M - __logf(S));
        cnt += 1.f;
    }
    sSum[threadIdx.x] = acc; sCnt[threadIdx.x] = cnt;
    __syncthreads();
    for (int o = 128; o > 0; o >>= 1) {
        if (threadIdx.x < o) {
            sSum[threadIdx.x] += sSum[threadIdx.x + o];
            sCnt[threadIdx.x] += sCnt[threadIdx.x + o];
        }
        __syncthreads();
    }
    if (threadIdx.x == 0) out[0] = sSum[0] / fmaxf(sCnt[0], 1.0f);
}

extern "C" void kernel_launch(void* const* d_in, const int* in_sizes, int n_in,
                              void* d_out, int out_size, void* d_ws, size_t ws_size,
                              hipStream_t stream) {
    (void)in_sizes; (void)n_in; (void)out_size; (void)ws_size;
    const float* hs     = (const float*)d_in[0];
    const float* w      = (const float*)d_in[1];
    const int*   labels = (const int*)d_in[2];
    float* out = (float*)d_out;

    char* ws = (char*)d_ws;
    _Float16* hs16 = (_Float16*)(ws + OFF_HS);
    _Float16* w16  = (_Float16*)(ws + OFF_W);
    float* pm  = (float*)(ws + OFF_PM);
    float* ps  = (float*)(ws + OFF_PS);
    float* lab = (float*)(ws + OFF_LB);

    ce_cvt_hidden<<<(NPAD*(H_/4) + 255)/256, 256, 0, stream>>>(hs, hs16);
    {
        long n4 = (long)V_ * (H_/4);
        ce_cvt_weight<<<(unsigned)((n4 + 255)/256), 256, 0, stream>>>(w, w16);
    }
    dim3 grid(NPAD/MBLK, NCHUNK);                      // x-fastest => chunk-major L2 reuse
    size_t shmem = APAN_BYTES + (size_t)MBLK*NWAVE*2*sizeof(float) + MBLK*sizeof(int);
    ce_gemm<<<grid, 512, shmem, stream>>>(hs16, w16, labels, pm, ps, lab);
    ce_final<<<1, 256, 0, stream>>>(pm, ps, lab, labels, out);
}
